// MultiHeadAttention_84997402788542
// MI455X (gfx1250) — matrix operand order repaired and checked by reading, not compiled, not run on hardware
//
#include <hip/hip_runtime.h>
#include <hip/hip_bf16.h>

// ---------------------------------------------------------------------------
// Sigmoid multi-head attention, CDNA5 (gfx1250), wave32 WMMA f16 path.
//   S=2048, D=1024, H=16, P=64, F=1024
// Stages:
//   1) fp32 -> f16 convert of x; transpose+convert of Qw/Kw/Vw ([h][d][p] ->
//      [h][p][d]) and W_fin ([k][n] -> [n][k]) so WMMA B-fragments are
//      contiguous loads.
//   2) QKV GEMMs: one wave -> 16(s) x 64(p) tile, explicitly double-buffered
//      (next k-step fragments loaded before current WMMAs issue).
//   3) Fused sigmoid attention, split-K over keys: 4 waves/block each own a
//      512-key quarter of the same (h, qtile); V/K fragments issued before
//      the score->sigmoid->LDS->relayout chain; LDS block reduction.
//   4) Final GEMM [S,1024]x[1024,1024] + bias, 16x64 tiles, double-buffered.
// ---------------------------------------------------------------------------

#define SEQ   2048
#define DDIM  1024
#define NH    16
#define PDIM  64
#define FDIM  1024

typedef __attribute__((ext_vector_type(16))) _Float16 v16h;
typedef __attribute__((ext_vector_type(8)))  _Float16 v8h;
typedef __attribute__((ext_vector_type(8)))  float    v8f;

// A-fragment (16x32 f16, lane holds row m=l&15): 16 contiguous K at 16*(lane>=16).
// `p` must already include the +16*hi lane offset.
static __device__ __forceinline__ v16h load_a_frag(const _Float16* p) {
  return *(const v16h*)(p);
}
// B-fragment (32x16 f16, lane holds col n=l&15): 16 contiguous K at 16*(lane>=16).
// `p` must already include the +16*hi lane offset.
static __device__ __forceinline__ v16h load_b_frag(const _Float16* p) {
  return *(const v16h*)(p);
}
static __device__ __forceinline__ v8f wmma_f16(v16h a, v16h b, v8f c) {
  return __builtin_amdgcn_wmma_f32_16x16x32_f16(false, a, false, b, (short)0, c,
                                                false, false);
}

// ---------------------------------------------------------------------------
// Stage 1: conversions / transposes
// ---------------------------------------------------------------------------
__global__ void k_cvt_f16(const float* __restrict__ src,
                          _Float16* __restrict__ dst, int n) {
  int i = blockIdx.x * 256 + threadIdx.x;
  if (i < n) dst[i] = (_Float16)src[i];
}

// [h][d][p] fp32 -> [h][p][d] f16   (H*P*D elements, D=1024, P=64)
__global__ void k_cvt_wT(const float* __restrict__ src,
                         _Float16* __restrict__ dst) {
  int i = blockIdx.x * 256 + threadIdx.x;        // over H*P*D
  int d = i & (DDIM - 1);
  int p = (i >> 10) & (PDIM - 1);
  int h = i >> 16;
  dst[i] = (_Float16)src[h * (DDIM * PDIM) + d * PDIM + p];
}

// [k][n] fp32 -> [n][k] f16   (1024x1024)
__global__ void k_cvt_wfT(const float* __restrict__ src,
                          _Float16* __restrict__ dst) {
  int i = blockIdx.x * 256 + threadIdx.x;        // over F*D
  int k = i & (DDIM - 1);
  int n = i >> 10;
  dst[i] = (_Float16)src[k * FDIM + n];
}

// ---------------------------------------------------------------------------
// Double-buffered 16x64 GEMM body: acc[0..3] += A(ap) * B(bp + j*16*ldb)
// over K = kdim. Next k-step's fragments are loaded before current WMMAs.
// ---------------------------------------------------------------------------
static __device__ __forceinline__ void gemm_16x64(
    const _Float16* __restrict__ ap, const _Float16* __restrict__ bp,
    int ldb, int kdim, v8f acc[4]) {
  v16h a_c  = load_a_frag(ap);
  v16h b_c0 = load_b_frag(bp + 0 * 16 * ldb);
  v16h b_c1 = load_b_frag(bp + 1 * 16 * ldb);
  v16h b_c2 = load_b_frag(bp + 2 * 16 * ldb);
  v16h b_c3 = load_b_frag(bp + 3 * 16 * ldb);
  for (int k0 = 0; k0 < kdim - 32; k0 += 32) {
    const int kn = k0 + 32;
    v16h a_n  = load_a_frag(ap + kn);
    v16h b_n0 = load_b_frag(bp + 0 * 16 * ldb + kn);
    v16h b_n1 = load_b_frag(bp + 1 * 16 * ldb + kn);
    v16h b_n2 = load_b_frag(bp + 2 * 16 * ldb + kn);
    v16h b_n3 = load_b_frag(bp + 3 * 16 * ldb + kn);
    acc[0] = wmma_f16(a_c, b_c0, acc[0]);
    acc[1] = wmma_f16(a_c, b_c1, acc[1]);
    acc[2] = wmma_f16(a_c, b_c2, acc[2]);
    acc[3] = wmma_f16(a_c, b_c3, acc[3]);
    a_c = a_n; b_c0 = b_n0; b_c1 = b_n1; b_c2 = b_n2; b_c3 = b_n3;
  }
  acc[0] = wmma_f16(a_c, b_c0, acc[0]);
  acc[1] = wmma_f16(a_c, b_c1, acc[1]);
  acc[2] = wmma_f16(a_c, b_c2, acc[2]);
  acc[3] = wmma_f16(a_c, b_c3, acc[3]);
}

// ---------------------------------------------------------------------------
// Stage 2: QKV projections. One wave -> 16(s) x 64(p) tile of Q, K or V.
// waves = 3 * H * (S/16) = 6144 ; 4 waves / block.
// ---------------------------------------------------------------------------
__global__ __launch_bounds__(128) void k_qkv(
    const _Float16* __restrict__ xh,    // [S][D]
    const _Float16* __restrict__ qwt,   // [H][P][D]
    const _Float16* __restrict__ kwt,
    const _Float16* __restrict__ vwt,
    _Float16* __restrict__ Qh,          // [H][S][P]
    _Float16* __restrict__ Kh,          // [H][S][P]
    _Float16* __restrict__ VT) {        // [H][P][S]
  const int wid = blockIdx.x * 4 + (threadIdx.x >> 5);
  const int mat = wid % 3;              // 0=Q 1=K 2=V
  const int w2  = wid / 3;
  const int qt  = w2 & 127;             // s tile (S/16 = 128)
  const int h   = w2 >> 7;

  const int l   = threadIdx.x & 31;
  const int r15 = l & 15;
  const int hi  = l >> 4;

  const _Float16* wt = (mat == 0) ? qwt : (mat == 1) ? kwt : vwt;
  const _Float16* ap = xh + (qt * 16 + r15) * DDIM + hi * 16;
  const _Float16* bp = wt + (h * PDIM + r15) * DDIM + hi * 16;

  v8f acc[4] = {};
  gemm_16x64(ap, bp, DDIM, DDIM, acc);

  if (mat < 2) {
    _Float16* dst = ((mat == 0) ? Qh : Kh) + h * SEQ * PDIM;
#pragma unroll
    for (int j = 0; j < 4; ++j)
#pragma unroll
      for (int r = 0; r < 8; ++r) {
        int s = qt * 16 + r + 8 * hi;
        dst[s * PDIM + j * 16 + r15] = (_Float16)acc[j][r];
      }
  } else {
    // V stored transposed: VT[h][p][s]; lane's 8 elems are contiguous in s.
#pragma unroll
    for (int j = 0; j < 4; ++j) {
      v8h pk;
#pragma unroll
      for (int r = 0; r < 8; ++r) pk[r] = (_Float16)acc[j][r];
      *(v8h*)(VT + (h * PDIM + j * 16 + r15) * SEQ + qt * 16 + 8 * hi) = pk;
    }
  }
}

// ---------------------------------------------------------------------------
// Stage 3: fused sigmoid attention, split over keys.
// One block (4 waves) owns one (h, qtile); wave w streams keys
// [w*512, w*512+512); partial 16x64 f32 accumulators reduced through LDS.
// blocks = H * (S/16) = 2048, 128 threads. 8192 waves on the hot stage.
// ---------------------------------------------------------------------------
__global__ __launch_bounds__(128) void k_attn(
    const _Float16* __restrict__ Qh,    // [H][S][P]
    const _Float16* __restrict__ Kh,    // [H][S][P]
    const _Float16* __restrict__ VT,    // [H][P][S]
    _Float16* __restrict__ cc) {        // concat [S][H*P]
  __shared__ __align__(32) _Float16 sc[4][16 * 32];   // per-wave score tile
  __shared__ float red[3][32][32];                    // [srcwave][elem][lane]

  const int h  = blockIdx.x >> 7;
  const int qt = blockIdx.x & 127;
  const int w  = threadIdx.x >> 5;      // wave in block: key quarter
  const int l  = threadIdx.x & 31;
  const int r15 = l & 15;
  const int hi  = l >> 4;

  _Float16* scw = sc[w];

  const _Float16* qrow = Qh + (h * SEQ + qt * 16 + r15) * PDIM + hi * 16;
  const v16h aq0 = load_a_frag(qrow);        // K-dim p = 0..31
  const v16h aq1 = load_a_frag(qrow + 32);   // K-dim p = 32..63

  const _Float16* kbase = Kh + h * SEQ * PDIM;
  const _Float16* krow  = kbase + r15 * PDIM + hi * 16;
  const _Float16* vrow  = VT + (h * PDIM + r15) * SEQ + hi * 16;

  v8f acc[4] = {};

  const int tBeg = w * (SEQ / 4);
  const int tEnd = tBeg + (SEQ / 4);
  for (int t0 = tBeg; t0 < tEnd; t0 += 32) {
    // Issue V fragments first: they only depend on t0 and have the whole
    // score -> sigmoid -> LDS round-trip to land.
    const _Float16* vp = vrow + t0;
    v16h bv0 = load_b_frag(vp + 0 * 16 * SEQ);
    v16h bv1 = load_b_frag(vp + 1 * 16 * SEQ);
    v16h bv2 = load_b_frag(vp + 2 * 16 * SEQ);
    v16h bv3 = load_b_frag(vp + 3 * 16 * SEQ);

    // Key fragments for both 16-wide score subtiles, batched.
    const _Float16* kp0 = krow + (t0 +  0) * PDIM;
    const _Float16* kp1 = krow + (t0 + 16) * PDIM;
    v16h bk00 = load_b_frag(kp0);
    v16h bk01 = load_b_frag(kp0 + 32);
    v16h bk10 = load_b_frag(kp1);
    v16h bk11 = load_b_frag(kp1 + 32);

    // prefetch next key block while this one computes
    __builtin_prefetch(kbase + (t0 + 32 + r15) * PDIM, 0, 1);

    // ---- score subtiles: sigmoid(Q . K^T / 64) -> f16 -> LDS [16][32] ----
    v8f c0 = {};
    c0 = wmma_f16(aq0, bk00, c0);
    c0 = wmma_f16(aq1, bk01, c0);
    v8f c1 = {};
    c1 = wmma_f16(aq0, bk10, c1);
    c1 = wmma_f16(aq1, bk11, c1);
#pragma unroll
    for (int r = 0; r < 8; ++r) {
      float e0  = __expf(c0[r] * -0.015625f);       // /64
      float sg0 = __builtin_amdgcn_rcpf(1.0f + e0);
      scw[(r + 8 * hi) * 32 + r15] = (_Float16)sg0;
      float e1  = __expf(c1[r] * -0.015625f);
      float sg1 = __builtin_amdgcn_rcpf(1.0f + e1);
      scw[(r + 8 * hi) * 32 + 16 + r15] = (_Float16)sg1;
    }

    // ---- relayout C -> A fragment via LDS, then score . V ----
    v16h as = load_a_frag(scw + r15 * 32 + hi * 16);
    acc[0] = wmma_f16(as, bv0, acc[0]);
    acc[1] = wmma_f16(as, bv1, acc[1]);
    acc[2] = wmma_f16(as, bv2, acc[2]);
    acc[3] = wmma_f16(as, bv3, acc[3]);
  }

  // ---- block reduction of partial accumulators ----
  if (w > 0) {
#pragma unroll
    for (int j = 0; j < 4; ++j)
#pragma unroll
      for (int r = 0; r < 8; ++r)
        red[w - 1][j * 8 + r][l] = acc[j][r];
  }
  __syncthreads();
  if (w == 0) {
#pragma unroll
    for (int j = 0; j < 4; ++j)
#pragma unroll
      for (int r = 0; r < 8; ++r)
        acc[j][r] += red[0][j * 8 + r][l] + red[1][j * 8 + r][l] +
                     red[2][j * 8 + r][l];

    // store into concat layout [s][h*64 + p] as f16 (A-side of final GEMM)
#pragma unroll
    for (int r = 0; r < 8; ++r) {
      int s = qt * 16 + r + 8 * hi;
      _Float16* row = cc + s * (NH * PDIM) + h * PDIM;
#pragma unroll
      for (int j = 0; j < 4; ++j)
        row[j * 16 + r15] = (_Float16)acc[j][r];
    }
  }
}

// ---------------------------------------------------------------------------
// Stage 4: final projection  out = concat @ W_fin + b   (fp32 out)
// One wave -> 16(m) x 64(n) tile, double-buffered.
// waves = (S/16)*(F/64) = 2048 ; 4 waves / block.
// ---------------------------------------------------------------------------
__global__ __launch_bounds__(128) void k_final(
    const _Float16* __restrict__ cc,    // [S][1024]
    const _Float16* __restrict__ wft,   // [N][K] = W_fin^T f16
    const float* __restrict__ bfin,     // [1024]
    float* __restrict__ out) {          // [S][1024] fp32
  const int wid = blockIdx.x * 4 + (threadIdx.x >> 5);
  const int nb  = wid & 15;             // 64-wide n block
  const int mt  = wid >> 4;
  const int l   = threadIdx.x & 31;
  const int r15 = l & 15;
  const int hi  = l >> 4;

  const _Float16* ap = cc  + (mt * 16 + r15) * (NH * PDIM) + hi * 16;
  const _Float16* bp = wft + (nb * 64 + r15) * FDIM + hi * 16;

  v8f acc[4] = {};
  gemm_16x64(ap, bp, FDIM, NH * PDIM, acc);

#pragma unroll
  for (int j = 0; j < 4; ++j) {
    const float bias = bfin[nb * 64 + j * 16 + r15];
#pragma unroll
    for (int r = 0; r < 8; ++r) {
      int s = mt * 16 + r + 8 * hi;
      out[s * FDIM + nb * 64 + j * 16 + r15] = acc[j][r] + bias;
    }
  }
}

// ---------------------------------------------------------------------------
// Host-side launch (graph-capture safe: only kernel launches on `stream`).
// Workspace layout (f16), total ~29.4 MB:
//   xh[S*D] | QwT[H*P*D] | KwT | VwT | WfT[F*D] | Qh[H*S*P] | Kh | VT | cc[S*H*P]
// ---------------------------------------------------------------------------
extern "C" void kernel_launch(void* const* d_in, const int* in_sizes, int n_in,
                              void* d_out, int out_size, void* d_ws, size_t ws_size,
                              hipStream_t stream) {
  const float* x    = (const float*)d_in[0];
  const float* Qw   = (const float*)d_in[1];
  const float* Kw   = (const float*)d_in[2];
  const float* Vw   = (const float*)d_in[3];
  const float* Wfin = (const float*)d_in[4];
  const float* bfin = (const float*)d_in[5];
  float* out = (float*)d_out;

  char* ws = (char*)d_ws;
  const size_t SZ_XH  = (size_t)SEQ * DDIM * 2;        // 4 MB
  const size_t SZ_WT  = (size_t)NH * PDIM * DDIM * 2;  // 2 MB
  const size_t SZ_WFT = (size_t)FDIM * DDIM * 2;       // 2 MB
  const size_t SZ_QKV = (size_t)NH * SEQ * PDIM * 2;   // 4 MB

  _Float16* xh  = (_Float16*)(ws);
  _Float16* qwt = (_Float16*)(ws + SZ_XH);
  _Float16* kwt = (_Float16*)(ws + SZ_XH + SZ_WT);
  _Float16* vwt = (_Float16*)(ws + SZ_XH + 2 * SZ_WT);
  _Float16* wft = (_Float16*)(ws + SZ_XH + 3 * SZ_WT);
  _Float16* Qh  = (_Float16*)(ws + SZ_XH + 3 * SZ_WT + SZ_WFT);
  _Float16* Kh  = (_Float16*)(ws + SZ_XH + 3 * SZ_WT + SZ_WFT + SZ_QKV);
  _Float16* VT  = (_Float16*)(ws + SZ_XH + 3 * SZ_WT + SZ_WFT + 2 * SZ_QKV);
  _Float16* cc  = (_Float16*)(ws + SZ_XH + 3 * SZ_WT + SZ_WFT + 3 * SZ_QKV);
  (void)ws_size; (void)in_sizes; (void)n_in; (void)out_size;

  // Stage 1: convert / transpose
  k_cvt_f16<<<(SEQ * DDIM) / 256, 256, 0, stream>>>(x, xh, SEQ * DDIM);
  k_cvt_wT <<<(NH * PDIM * DDIM) / 256, 256, 0, stream>>>(Qw, qwt);
  k_cvt_wT <<<(NH * PDIM * DDIM) / 256, 256, 0, stream>>>(Kw, kwt);
  k_cvt_wT <<<(NH * PDIM * DDIM) / 256, 256, 0, stream>>>(Vw, vwt);
  k_cvt_wfT<<<(FDIM * DDIM) / 256, 256, 0, stream>>>(Wfin, wft);

  // Stage 2: QKV projections (6144 waves, 16x64 tiles)
  k_qkv<<<(3 * NH * (SEQ / 16)) / 4, 128, 0, stream>>>(
      xh, qwt, kwt, vwt, Qh, Kh, VT);

  // Stage 3: fused sigmoid attention (2048 blocks, 8192 waves, split-K)
  k_attn<<<NH * (SEQ / 16), 128, 0, stream>>>(Qh, Kh, VT, cc);

  // Stage 4: final projection (2048 waves, 16x64 tiles)
  k_final<<<((SEQ / 16) * (FDIM / 64)) / 4, 128, 0, stream>>>(cc, wft, bfin, out);
}
